// FERNN_Cell_46806553592094
// MI455X (gfx1250) — matrix operand order repaired and checked by reading, not compiled, not run on hardware
//
#include <hip/hip_runtime.h>

// ---------------------------------------------------------------------------
// FERNN cell, CDNA5 (gfx1250) implicit-GEMM with V_WMMA_F32_16X16X32_BF16.
//   out = relu( pad(circ_conv(u_t, w_u)) + circ_conv(roll(h_prev, action), w_h) )
// B=32, CIN=3, CH=128, WIN=64, WORLD=128, K=3
// ---------------------------------------------------------------------------

typedef __attribute__((ext_vector_type(16))) __bf16 v16bf;
typedef __attribute__((ext_vector_type(8)))  __bf16 v8bf;
typedef __attribute__((ext_vector_type(8)))  float  v8f;

#define BATCH  32
#define CIN    3
#define CH     128
#define WIN    64
#define WORLD  128
#define ICP    136          // LDS inner pitch: 128 channels + 8 pad (bank rotate)
#define NCHUNK 36           // 9 taps * 128 ic / K32

// ---- prep: repack w_h [oc][ic][ky][kx] f32 -> [tap][oc][ic] bf16 -----------
__global__ void pack_wh(const float* __restrict__ wh, __bf16* __restrict__ dst) {
    int i = blockIdx.x * blockDim.x + threadIdx.x;
    if (i >= 9 * CH * CH) return;
    int ic = i & 127;
    int oc = (i >> 7) & 127;
    int t  = i >> 14;                       // tap = ky*3+kx
    dst[i] = (__bf16)wh[oc * (CH * 9) + ic * 9 + t];
}

// ---- prep: repack w_u [oc][cin][ky][kx] f32 -> [oc][32] bf16 (K=27 pad 32) -
__global__ void pack_wu(const float* __restrict__ wu, __bf16* __restrict__ dst) {
    int i = blockIdx.x * blockDim.x + threadIdx.x;
    if (i >= CH * 32) return;
    int k  = i & 31;                        // k = tap*3 + cin
    int oc = i >> 5;
    float v = 0.f;
    if (k < 27) {
        int cin = k % 3, t = k / 3;
        v = wu[oc * 27 + cin * 9 + t];
    }
    dst[i] = (__bf16)v;
}

// ---- main fused kernel ------------------------------------------------------
__launch_bounds__(256, 1)
__global__ void fernn_cell(const float* __restrict__ h_prev,
                           const float* __restrict__ u_t,
                           const int*   __restrict__ act,
                           const __bf16* __restrict__ whp,
                           const __bf16* __restrict__ wup,
                           float* __restrict__ out) {
    __shared__ __bf16 lds_h[3 * 66 * ICP];   // 3 rows x 66 cols x 128ch halo (bf16)
    __shared__ __bf16 u_im2col[64 * 32];     // 64 pixels x K32 patch for u_conv

    const int x0  = blockIdx.x * 64;         // pixel-column block (0 or 64)
    const int y   = blockIdx.y;              // output row
    const int b   = blockIdx.z;              // batch
    const int tid = threadIdx.x;

    const int dx = act[2 * b + 0];
    const int dy = act[2 * b + 1];

    // ---- stage shifted h halo: rows y-1..y+1, cols x0-1..x0+64, fp32->bf16 --
    const float* hb = h_prev + (size_t)b * CH * WORLD * WORLD;
    for (int e = tid; e < 3 * 66 * CH; e += 256) {
        int ic = e & 127;
        int rc = e >> 7;
        int c  = rc % 66;
        int r  = rc / 66;
        int gr = (y - 1 + r + dy + WORLD) & (WORLD - 1);   // roll folded in
        int gc = (x0 - 1 + c + dx + WORLD) & (WORLD - 1);
        float v = hb[((size_t)ic * WORLD + gr) * WORLD + gc];
        lds_h[(r * 66 + c) * ICP + ic] = (__bf16)v;
    }

    // ---- stage u_conv im2col patch (only where window overlaps this tile) ---
    const bool do_u = (x0 == 0) && (y < WIN);
    if (do_u) {
        const float* ub = u_t + (size_t)b * CIN * WIN * WIN;
        for (int e = tid; e < 64 * 32; e += 256) {
            int k = e & 31;                  // k = tap*3 + cin  (27..31 = zero pad)
            int j = e >> 5;                  // output pixel x (== global x here)
            float v = 0.f;
            if (k < 27) {
                int cin = k % 3, t = k / 3, ky = t / 3, kx = t % 3;
                int rr = (y + ky - 1 + WIN) & (WIN - 1);   // circular on 64x64
                int cc = (j + kx - 1 + WIN) & (WIN - 1);
                v = ub[((size_t)cin * WIN + rr) * WIN + cc];
            }
            u_im2col[e] = (__bf16)v;
        }
    }
    __syncthreads();

    const int wave  = tid >> 5;              // 8 waves -> 8 M-tiles of 16 oc
    const int lane  = tid & 31;
    const int khalf = lane >> 4;             // K-half select per WMMA A/B layout
    const int l16   = lane & 15;
    const int oc0   = wave * 16;

    v8f acc[4];
    const v8f vzero = {0.f, 0.f, 0.f, 0.f, 0.f, 0.f, 0.f, 0.f};
    acc[0] = vzero; acc[1] = vzero; acc[2] = vzero; acc[3] = vzero;

    // ---- K loop: 36 chunks of K=32 (tap-major, 32 in-channels per chunk) ----
    for (int chunk = 0; chunk < NCHUNK; ++chunk) {
        const int tap = chunk >> 2;
        const int icq = (chunk & 3) << 5;
        const int ky  = tap / 3;
        const int kx  = tap % 3;

        // A fragment (weights): lanes 0-15 K{0..7,16..23}, lanes 16-31 K{8..15,24..31}
        const __bf16* ap = whp + (((size_t)tap * CH + (oc0 + l16)) * CH + icq + khalf * 8);
        v8bf alo = *(const v8bf*)ap;             // 16B contiguous
        v8bf ahi = *(const v8bf*)(ap + 16);      // +16 channels
        v16bf a  = __builtin_shufflevector(alo, ahi,
                     0,1,2,3,4,5,6,7,8,9,10,11,12,13,14,15);

#pragma unroll
        for (int n = 0; n < 4; ++n) {
            const int j = n * 16 + l16;          // output pixel (N = lane%16)
            // B fragment: lanes 0-15 K{0..7,16..23}, lanes 16-31 K{8..15,24..31}
            const __bf16* bp = lds_h + ((ky * 66 + (j + kx)) * ICP + icq + khalf * 8);
            v8bf blo = *(const v8bf*)bp;
            v8bf bhi = *(const v8bf*)(bp + 16);
            v16bf bm = __builtin_shufflevector(blo, bhi,
                         0,1,2,3,4,5,6,7,8,9,10,11,12,13,14,15);
            acc[n] = __builtin_amdgcn_wmma_f32_16x16x32_bf16(
                         false, a, false, bm, (short)0, acc[n], false, false);
        }
    }

    // ---- extra K=32 chunk: encoder conv contribution inside the window ------
    if (do_u) {
        const __bf16* ap = wup + ((size_t)(oc0 + l16) * 32 + khalf * 8);
        v8bf alo = *(const v8bf*)ap;
        v8bf ahi = *(const v8bf*)(ap + 16);
        v16bf a  = __builtin_shufflevector(alo, ahi,
                     0,1,2,3,4,5,6,7,8,9,10,11,12,13,14,15);
#pragma unroll
        for (int n = 0; n < 4; ++n) {
            const int j = n * 16 + l16;
            const __bf16* bp = u_im2col + (j * 32 + khalf * 8);
            v8bf blo = *(const v8bf*)bp;
            v8bf bhi = *(const v8bf*)(bp + 16);
            v16bf bm = __builtin_shufflevector(blo, bhi,
                         0,1,2,3,4,5,6,7,8,9,10,11,12,13,14,15);
            acc[n] = __builtin_amdgcn_wmma_f32_16x16x32_bf16(
                         false, a, false, bm, (short)0, acc[n], false, false);
        }
    }

    // ---- epilogue: ReLU + store f32 (C/D layout: VGPR r -> M=r+8*khalf) -----
    const size_t obase = (size_t)b * (CH * WORLD * WORLD) + (size_t)y * WORLD;
#pragma unroll
    for (int n = 0; n < 4; ++n) {
        const int x = x0 + n * 16 + l16;
#pragma unroll
        for (int r = 0; r < 8; ++r) {
            const int oc = oc0 + khalf * 8 + r;
            float v = acc[n][r];
            out[obase + (size_t)oc * (WORLD * WORLD) + x] = v > 0.f ? v : 0.f;
        }
    }
}

// ---------------------------------------------------------------------------
extern "C" void kernel_launch(void* const* d_in, const int* in_sizes, int n_in,
                              void* d_out, int out_size, void* d_ws, size_t ws_size,
                              hipStream_t stream) {
    const float* u_t    = (const float*)d_in[0];   // [32,3,64,64]
    const float* h_prev = (const float*)d_in[1];   // [32,128,128,128]
    const int*   action = (const int*)d_in[2];     // [32,2] (dx,dy)
    const float* w_u    = (const float*)d_in[3];   // [128,3,3,3]
    const float* w_h    = (const float*)d_in[4];   // [128,128,3,3]
    float* out = (float*)d_out;                    // [32,128,128,128]

    __bf16* whp = (__bf16*)d_ws;                   // 9*128*128 bf16
    __bf16* wup = whp + 9 * CH * CH;               // 128*32 bf16

    pack_wh<<<(9 * CH * CH + 255) / 256, 256, 0, stream>>>(w_h, whp);
    pack_wu<<<(CH * 32 + 255) / 256, 256, 0, stream>>>(w_u, wup);

    dim3 grid(WORLD / 64, WORLD, BATCH);           // (2, 128, 32)
    fernn_cell<<<grid, 256, 0, stream>>>(h_prev, u_t, action, whp, wup, out);

    (void)in_sizes; (void)n_in; (void)out_size; (void)ws_size;
}
